// CausalFlowModel_34213709480661
// MI455X (gfx1250) — matrix operand order repaired and checked by reading, not compiled, not run on hardware
//
#include <hip/hip_runtime.h>
#include <hip/hip_bf16.h>

// Problem constants (match reference)
#define B_    512
#define T_    1024
#define SD_   8
#define CD1   5      // CD + 1
#define Z_    72     // state size
#define G4    288    // 4*Z
#define KPAD  104    // padded K row stride (96 used, 104 for bank spread)
#define ENC_H_ 128
#define CRS_  64
#define DEC_H_ 144
#define OUT_  8

typedef __attribute__((ext_vector_type(16))) _Float16 v16h;
typedef __attribute__((ext_vector_type(8)))  _Float16 v8h;
typedef __attribute__((ext_vector_type(8)))  float    v8f;

union V16U { v16h v; v8h h[2]; };

// Fast transcendentals: gfx1250 has a hardware v_tanh_f32 (TRANS32), which
// co-executes with regular VALU.  sigmoid(x) = 0.5*tanh(x/2) + 0.5 turns each
// gate activation into one TRANS op + one FMA (no IEEE divide sequences).
#if __has_builtin(__builtin_amdgcn_tanhf)
__device__ __forceinline__ float ftanh(float x) { return __builtin_amdgcn_tanhf(x); }
#elif __has_builtin(__builtin_amdgcn_rcpf)
__device__ __forceinline__ float ftanh(float x) {
  return 1.0f - 2.0f * __builtin_amdgcn_rcpf(__expf(2.0f * x) + 1.0f);
}
#else
__device__ __forceinline__ float ftanh(float x) {
  return 1.0f - 2.0f / (__expf(2.0f * x) + 1.0f);
}
#endif
__device__ __forceinline__ float fsig(float x) { return fmaf(0.5f, ftanh(0.5f * x), 0.5f); }

// ---------------------------------------------------------------------------
// Encoder: z0[b] = [x[b], mlp3(x[b])]   (one wave per batch row)
// ---------------------------------------------------------------------------
__global__ void __launch_bounds__(256) enc_kernel(
    const float* __restrict__ x,
    const float* __restrict__ We1, const float* __restrict__ be1,
    const float* __restrict__ We2, const float* __restrict__ be2,
    const float* __restrict__ We3, const float* __restrict__ be3,
    float* __restrict__ z0)
{
  __shared__ float sv[8][ENC_H_];
  const int wave = threadIdx.x >> 5;
  const int lane = threadIdx.x & 31;
  const int b = blockIdx.x * 8 + wave;

  float xr[SD_];
#pragma unroll
  for (int k = 0; k < SD_; ++k) xr[k] = x[b * SD_ + k];

  // layer 1: 8 -> 128, tanh
  float v1[4];
#pragma unroll
  for (int c = 0; c < 4; ++c) {
    int col = lane + 32 * c;
    float acc = be1[col];
#pragma unroll
    for (int k = 0; k < SD_; ++k) acc += xr[k] * We1[k * ENC_H_ + col];
    v1[c] = ftanh(acc);
  }
#pragma unroll
  for (int c = 0; c < 4; ++c) sv[wave][lane + 32 * c] = v1[c];
  __syncthreads();

  // layer 2: 128 -> 128, tanh
  float v2[4];
#pragma unroll
  for (int c = 0; c < 4; ++c) {
    int col = lane + 32 * c;
    float acc = be2[col];
    for (int k = 0; k < ENC_H_; ++k) acc += sv[wave][k] * We2[k * ENC_H_ + col];
    v2[c] = ftanh(acc);
  }
  __syncthreads();
#pragma unroll
  for (int c = 0; c < 4; ++c) sv[wave][lane + 32 * c] = v2[c];
  __syncthreads();

  // layer 3: 128 -> 64
#pragma unroll
  for (int c = 0; c < 2; ++c) {
    int col = lane + 32 * c;
    float acc = be3[col];
    for (int k = 0; k < ENC_H_; ++k) acc += sv[wave][k] * We3[k * CRS_ + col];
    z0[b * Z_ + SD_ + col] = acc;
  }
  if (lane < SD_) z0[b * Z_ + lane] = xr[lane];
}

// ---------------------------------------------------------------------------
// Main LSTM scan. 16 batch rows per block, 9 wave32s, 18 N-tiles of 16 cols.
// gates = [h | u_t] @ [Whh ; Wih] + (bih+bhh) via 3x v_wmma_f32_16x16x32_f16
// per N-tile; weight (B) fragments stay resident in VGPRs for all T steps.
// ---------------------------------------------------------------------------
__global__ void __launch_bounds__(288) scan_kernel(
    const float* __restrict__ rnn_input,  // [B, T, 5]
    const float* __restrict__ deltas,     // [B, T]
    const int*   __restrict__ lengths,    // [B]
    const float* __restrict__ Wih,        // [5, 288]
    const float* __restrict__ Whh,        // [72, 288]
    const float* __restrict__ bih,        // [288]
    const float* __restrict__ bhh,        // [288]
    const float* __restrict__ z0,         // [B, 72]
    float* __restrict__ sel)              // [B, 72]
{
  __shared__ alignas(16) _Float16 sA[16 * KPAD];   // A staging: [h | u | 0pad]
  __shared__ float sG[16 * G4];                    // raw gates
  __shared__ float sD[16];                         // per-row delta_t
  __shared__ int   sIdx[16];                       // selection index per row
  __shared__ int   sTmax;

  const int tid   = threadIdx.x;
  const int lane  = tid & 31;
  const int wave  = tid >> 5;
  const int bBase = blockIdx.x * 16;
  const int l15   = lane & 15;
  const bool hiHalf = (lane >= 16);

  // zero the A staging buffer (K pad columns must stay 0 forever)
  for (int j = tid; j < 16 * KPAD; j += 288) sA[j] = (_Float16)0.0f;
  if (tid < 16) {
    int idx = lengths[bBase + tid] - 1;
    idx = idx < 0 ? 0 : (idx > T_ - 1 ? T_ - 1 : idx);
    sIdx[tid] = idx;
  }
  __syncthreads();
  if (tid == 0) {
    int m = 0;
    for (int r = 0; r < 16; ++r) m = m > sIdx[r] ? m : sIdx[r];
    sTmax = m + 1;
  }

  // per-thread recurrence state: 4 of the 16*72 (row,z) elements
  float hst[4], cst[4];
  int rowq[4], zq[4];
#pragma unroll
  for (int q = 0; q < 4; ++q) {
    int e = q * 288 + tid;
    rowq[q] = e / Z_;
    zq[q]   = e - rowq[q] * Z_;
    hst[q]  = z0[(bBase + rowq[q]) * Z_ + zq[q]];
    cst[q]  = 0.0f;
    sA[rowq[q] * KPAD + zq[q]] = (_Float16)hst[q];
  }

  // Preload stationary B fragments (2 n-tiles x 3 k-tiles) and bias accs.
  // B k order used here (contiguous, same as A): lanes 0-15 -> N=l, K=0..15; lanes 16-31 -> K=16..31.
  V16U bfrag[2][3];
  v8f  biasAcc[2];
#pragma unroll
  for (int tt = 0; tt < 2; ++tt) {
    const int ntile = wave * 2 + tt;
    const int n = ntile * 16 + l15;
    const float bv = bih[n] + bhh[n];
#pragma unroll
    for (int r = 0; r < 8; ++r) biasAcc[tt][r] = bv;
#pragma unroll
    for (int kt = 0; kt < 3; ++kt) {
#pragma unroll
      for (int j = 0; j < 16; ++j) {
        int k = kt * 32 + (hiHalf ? 16 : 0) + j;
        float w = 0.0f;
        if (k < Z_)            w = Whh[k * G4 + n];
        else if (k < Z_ + CD1) w = Wih[(k - Z_) * G4 + n];
        if (j < 8) bfrag[tt][kt].h[0][j]     = (_Float16)w;
        else       bfrag[tt][kt].h[1][j - 8] = (_Float16)w;
      }
    }
  }
  __syncthreads();
  const int tmax = sTmax;

  // A k order used here (contiguous, same as B): lanes 0-15: M=l, K=0..15;
  //                           lanes 16-31: M=l-16, K=16..31.
  const int aRow = l15;
  const int aK0  = hiHalf ? 16 : 0;

  for (int t = 0; t < tmax; ++t) {
    // stage u_t (16 rows x 5) and delta_t (16) into LDS
    if (tid < 80) {
      int row = tid / CD1, k = tid - row * CD1;
      sA[row * KPAD + Z_ + k] =
          (_Float16)rnn_input[((size_t)(bBase + row) * T_ + t) * CD1 + k];
    }
    if (tid >= 96 && tid < 112) {
      int row = tid - 96;
      sD[row] = deltas[(size_t)(bBase + row) * T_ + t];
      if (t + 1 < T_)
        __builtin_prefetch(&rnn_input[((size_t)(bBase + row) * T_ + (t + 1)) * CD1], 0, 3);
    }
    __syncthreads();

    // gather A fragments from LDS (2x ds_load_b128 per k-tile)
    V16U afrag[3];
#pragma unroll
    for (int kt = 0; kt < 3; ++kt) {
      const _Float16* base = &sA[aRow * KPAD + kt * 32 + aK0];
      afrag[kt].h[0] = *(const v8h*)(base);
      afrag[kt].h[1] = *(const v8h*)(base + 8);
    }

    // 2 N-tiles x 3 chained WMMAs (K = 96 covers 72 h + 5 u + zero pad)
#pragma unroll
    for (int tt = 0; tt < 2; ++tt) {
      v8f acc = biasAcc[tt];
#pragma unroll
      for (int kt = 0; kt < 3; ++kt) {
        acc = __builtin_amdgcn_wmma_f32_16x16x32_f16(
            false, afrag[kt].v, false, bfrag[tt][kt].v,
            (short)0, acc, false, false);
      }
      const int ncol  = (wave * 2 + tt) * 16 + l15;
      const int mBase = hiHalf ? 8 : 0;
#pragma unroll
      for (int r = 0; r < 8; ++r) sG[(mBase + r) * G4 + ncol] = acc[r];
    }
    __syncthreads();

    // activation + forward-Euler update (f32), capture selected timestep
#pragma unroll
    for (int q = 0; q < 4; ++q) {
      const int row = rowq[q], z = zq[q];
      const float* gr = &sG[row * G4];
      const float gi = fsig(gr[z]);
      const float gf = fsig(gr[Z_ + z]);
      const float gg = ftanh(gr[2 * Z_ + z]);
      const float go = fsig(gr[3 * Z_ + z]);
      const float cc = gf * cst[q] + gi * gg;
      const float hc = go * ftanh(cc);
      const float d  = sD[row];
      const float cn = cst[q] + d * (cc - cst[q]);
      const float hn = hst[q] + d * (hc - hst[q]);
      if (t == sIdx[row])
        sel[(size_t)(bBase + row) * Z_ + z] = (1.0f - d) * hst[q] + d * hn;
      cst[q] = cn;
      hst[q] = hn;
      sA[row * KPAD + z] = (_Float16)hn;
    }
    __syncthreads();
  }
}

// ---------------------------------------------------------------------------
// Decoder: out[b] = mlp3(sel[b])[:8]   (one wave per batch row)
// ---------------------------------------------------------------------------
__global__ void __launch_bounds__(256) dec_kernel(
    const float* __restrict__ selin,  // [B, 72]
    const float* __restrict__ Wd1, const float* __restrict__ bd1,
    const float* __restrict__ Wd2, const float* __restrict__ bd2,
    const float* __restrict__ Wd3, const float* __restrict__ bd3,
    float* __restrict__ out)          // [B, 8]
{
  __shared__ float sx[8][Z_];
  __shared__ float sv[8][DEC_H_];
  const int wave = threadIdx.x >> 5;
  const int lane = threadIdx.x & 31;
  const int b = blockIdx.x * 8 + wave;

  for (int k = lane; k < Z_; k += 32) sx[wave][k] = selin[b * Z_ + k];
  __syncthreads();

  float v1[5];
  int nc = 0;
  for (int col = lane; col < DEC_H_; col += 32) {
    float acc = bd1[col];
    for (int k = 0; k < Z_; ++k) acc += sx[wave][k] * Wd1[k * DEC_H_ + col];
    v1[nc++] = ftanh(acc);
  }
  nc = 0;
  for (int col = lane; col < DEC_H_; col += 32) sv[wave][col] = v1[nc++];
  __syncthreads();

  float v2[5];
  nc = 0;
  for (int col = lane; col < DEC_H_; col += 32) {
    float acc = bd2[col];
    for (int k = 0; k < DEC_H_; ++k) acc += sv[wave][k] * Wd2[k * DEC_H_ + col];
    v2[nc++] = ftanh(acc);
  }
  __syncthreads();
  nc = 0;
  for (int col = lane; col < DEC_H_; col += 32) sv[wave][col] = v2[nc++];
  __syncthreads();

  if (lane < OUT_) {
    float acc = bd3[lane];
    for (int k = 0; k < DEC_H_; ++k) acc += sv[wave][k] * Wd3[k * OUT_ + lane];
    out[b * OUT_ + lane] = acc;
  }
}

// ---------------------------------------------------------------------------
extern "C" void kernel_launch(void* const* d_in, const int* in_sizes, int n_in,
                              void* d_out, int out_size, void* d_ws, size_t ws_size,
                              hipStream_t stream) {
  (void)in_sizes; (void)n_in; (void)out_size; (void)ws_size;
  const float* x    = (const float*)d_in[0];
  const float* rnn  = (const float*)d_in[1];
  const float* del  = (const float*)d_in[2];
  const int*   len  = (const int*)d_in[3];
  const float* We1  = (const float*)d_in[4];
  const float* be1  = (const float*)d_in[5];
  const float* We2  = (const float*)d_in[6];
  const float* be2  = (const float*)d_in[7];
  const float* We3  = (const float*)d_in[8];
  const float* be3  = (const float*)d_in[9];
  const float* Wih  = (const float*)d_in[10];
  const float* Whh  = (const float*)d_in[11];
  const float* bih  = (const float*)d_in[12];
  const float* bhh  = (const float*)d_in[13];
  const float* Wd1  = (const float*)d_in[14];
  const float* bd1  = (const float*)d_in[15];
  const float* Wd2  = (const float*)d_in[16];
  const float* bd2  = (const float*)d_in[17];
  const float* Wd3  = (const float*)d_in[18];
  const float* bd3  = (const float*)d_in[19];
  float* out = (float*)d_out;

  float* z0  = (float*)d_ws;          // [B, 72]
  float* sel = z0 + (size_t)B_ * Z_;  // [B, 72]

  enc_kernel<<<B_ / 8, 256, 0, stream>>>(x, We1, be1, We2, be2, We3, be3, z0);
  scan_kernel<<<B_ / 16, 288, 0, stream>>>(rnn, del, len, Wih, Whh, bih, bhh, z0, sel);
  dec_kernel<<<B_ / 8, 256, 0, stream>>>(sel, Wd1, bd1, Wd2, bd2, Wd3, bd3, out);
}
